// MaxKSAGEConv_11768210391445
// MI455X (gfx1250) — matrix operand order repaired and checked by reading, not compiled, not run on hardware
//
#include <hip/hip_runtime.h>
#include <hip/hip_bf16.h>
#include <stdint.h>

// ---------------------------------------------------------------------------
// MaxK-SAGE conv for MI455X (gfx1250, wave32).
//   h_self  = feat @ W_self          (bf16 WMMA, f32 accum)
//   h_neigh = feat @ W_neigh         (fused in same kernel: shared A frags)
//   h_sparse = top-32-per-row(h_neigh)   -> compressed (vals, cols)
//   out = h_self + segment_mean(h_sparse[src], dst)
// ---------------------------------------------------------------------------

typedef __bf16 bf16_t;
typedef bf16_t v16bf __attribute__((ext_vector_type(16)));
typedef bf16_t v8bf  __attribute__((ext_vector_type(8)));
typedef float  v8f   __attribute__((ext_vector_type(8)));
typedef float  v4f   __attribute__((ext_vector_type(4)));

#define KDIM 256
#define NOUT 256
#define TOPK 32

// ---------------- convert feat f32 -> bf16 (8 elems/thread) ----------------
__global__ void cvt_feat8(const float* __restrict__ f, bf16_t* __restrict__ o,
                          long long n8) {
  long long t = (long long)blockIdx.x * blockDim.x + threadIdx.x;
  if (t >= n8) return;
  const float4* fp = (const float4*)f;
  float4 a = fp[2 * t];
  float4 b = fp[2 * t + 1];
  v8bf r = {(bf16_t)a.x, (bf16_t)a.y, (bf16_t)a.z, (bf16_t)a.w,
            (bf16_t)b.x, (bf16_t)b.y, (bf16_t)b.z, (bf16_t)b.w};
  ((v8bf*)o)[t] = r;
}

// -------- convert + transpose W [K,N] f32 -> Wt [N,K] bf16 ------------------
__global__ void cvt_w_transpose(const float* __restrict__ w,
                                bf16_t* __restrict__ wt) {
  int t = blockIdx.x * blockDim.x + threadIdx.x;   // 0..65535
  int k = t >> 8;
  int n = t & 255;
  wt[n * KDIM + k] = (bf16_t)w[t];
}

// ---------------- fused dual GEMM: Cs = A*WsT', Cn = A*WnT' -----------------
// Each wave computes a 16(M) x 64(N) strip of BOTH outputs.
// A frag (16-bit A 16x32, same K order as B): lanes 0-15 -> M=lane, K 0..15;
//                                      lanes16-31 -> M=lane-16, K 16..31
// B frag (32x16): lanes 0-15 -> N=lane, K 0..15; lanes 16-31 -> N, K 16..31.
// W is pre-transposed to [N,K] so a B column is a contiguous 32B chunk.
__global__ __launch_bounds__(256) void gemm_dual(
    const bf16_t* __restrict__ A,    // [M,256]
    const bf16_t* __restrict__ BtS,  // [256,256] = W_self^T
    const bf16_t* __restrict__ BtN,  // [256,256] = W_neigh^T
    float* __restrict__ Cs,          // [M,256]  (d_out: h_self)
    float* __restrict__ Cn,          // [M,256]  (ws: h_neigh)
    int M, int Mtiles) {
  int wave = blockIdx.x * (blockDim.x >> 5) + (threadIdx.x >> 5);
  int mt = wave >> 2;        // M tile index
  int ns = wave & 3;         // N strip (64 cols each)
  if (mt >= Mtiles) return;
  int lane = threadIdx.x & 31;
  int half = lane >> 4;
  int lidx = lane & 15;
  int m0 = mt * 16;
  int n0 = ns * 64;

  int arow_i = m0 + lidx;
  if (arow_i >= M) arow_i = M - 1;                 // tail clamp (loads only)
  const bf16_t* arow = A + (size_t)arow_i * KDIM + half * 16;

  v8f accS[4], accN[4];
  v8f vzero = {0.f, 0.f, 0.f, 0.f, 0.f, 0.f, 0.f, 0.f};
#pragma unroll
  for (int j = 0; j < 4; ++j) { accS[j] = vzero; accN[j] = vzero; }

  for (int k0 = 0; k0 < KDIM; k0 += 32) {
    v8bf alo = *(const v8bf*)(arow + k0);        // K {k0+0..7 | k0+16..23}
    v8bf ahi = *(const v8bf*)(arow + k0 + 8);    // K {k0+8..15 | k0+24..31}
    v16bf a = __builtin_shufflevector(alo, ahi, 0, 1, 2, 3, 4, 5, 6, 7, 8, 9,
                                      10, 11, 12, 13, 14, 15);
#pragma unroll
    for (int j = 0; j < 4; ++j) {
      size_t boff = (size_t)(n0 + j * 16 + lidx) * KDIM + k0 + half * 16;
      v16bf bs = *(const v16bf*)(BtS + boff);    // contiguous K halves
      accS[j] = __builtin_amdgcn_wmma_f32_16x16x32_bf16(
          false, a, false, bs, (short)0, accS[j], false, false);
      v16bf bn = *(const v16bf*)(BtN + boff);
      accN[j] = __builtin_amdgcn_wmma_f32_16x16x32_bf16(
          false, a, false, bn, (short)0, accN[j], false, false);
    }
  }

  // C/D layout: lane<16 -> N=lane, M = vgpr i (0..7); lane>=16 -> M = 8+i.
#pragma unroll
  for (int j = 0; j < 4; ++j) {
#pragma unroll
    for (int i = 0; i < 8; ++i) {
      int m = m0 + half * 8 + i;
      if (m < M) {
        size_t o = (size_t)m * NOUT + n0 + j * 16 + lidx;
        Cs[o] = accS[j][i];
        Cn[o] = accN[j][i];
      }
    }
  }
}

// ---------------- MaxK: top-32 per 256-wide row, emit sparse ----------------
// One wave per row; lane holds cols [lane*8, lane*8+7]. After extracting the
// sparse row, the dense row is zeroed so the buffer doubles as the segment-sum
// accumulator (stays resident in the 192MB L2).
__global__ __launch_bounds__(256) void maxk_sparse(
    float* __restrict__ h,          // [N,256] read, then zeroed
    float* __restrict__ vals,       // [N,32]
    int* __restrict__ cols,         // [N,32]
    int nrows) {
  int row = blockIdx.x * (blockDim.x >> 5) + (threadIdx.x >> 5);
  if (row >= nrows) return;
  int lane = threadIdx.x & 31;
  float* rp = h + (size_t)row * NOUT;
  v4f a = *(const v4f*)(rp + lane * 8);
  v4f b = *(const v4f*)(rp + lane * 8 + 4);
  float v[8] = {a.x, a.y, a.z, a.w, b.x, b.y, b.z, b.w};
  unsigned rem = 0;
  for (int t = 0; t < TOPK; ++t) {
    float lm = -3.402823466e38f;
    int li = 0;
#pragma unroll
    for (int i = 0; i < 8; ++i) {
      bool ok = ((rem >> i) & 1u) == 0u && v[i] > lm;
      if (ok) { lm = v[i]; li = i; }
    }
    float wm = lm;
#pragma unroll
    for (int off = 16; off > 0; off >>= 1)
      wm = fmaxf(wm, __shfl_xor(wm, off, 32));
    unsigned long long bal = __ballot(lm == wm);
    int winner = __ffsll(bal) - 1;
    if (lane == winner) {
      rem |= (1u << li);
      vals[(size_t)row * TOPK + t] = lm;
      cols[(size_t)row * TOPK + t] = lane * 8 + li;
    }
  }
  v4f z = {0.f, 0.f, 0.f, 0.f};
  *(v4f*)(rp + lane * 8) = z;
  *(v4f*)(rp + lane * 8 + 4) = z;
}

// ---------------- zero helper ----------------------------------------------
__global__ void zero_f32(float* __restrict__ p, int n) {
  int t = blockIdx.x * blockDim.x + threadIdx.x;
  if (t < n) p[t] = 0.f;
}

// ---------------- edge scatter: one wave per edge ---------------------------
// 32 lanes: one (val,col) pair each -> atomic f32 add into accum[dst].
__global__ __launch_bounds__(256) void edge_scatter(
    const int* __restrict__ src, const int* __restrict__ dst,
    const float* __restrict__ vals, const int* __restrict__ cols,
    float* __restrict__ accum, float* __restrict__ deg, int nedges) {
  int e = blockIdx.x * (blockDim.x >> 5) + (threadIdx.x >> 5);
  if (e >= nedges) return;
  int lane = threadIdx.x & 31;
  int s = src[e];
  int d = dst[e];
  float v = vals[(size_t)s * TOPK + lane];
  int c = cols[(size_t)s * TOPK + lane];
  atomicAdd(&accum[(size_t)d * NOUT + c], v);
  if (lane == 0) atomicAdd(&deg[d], 1.0f);
}

// ---------------- finalize: out = h_self + accum / max(deg,1) --------------
__global__ void finalize_mean(float* __restrict__ out,
                              const float* __restrict__ accum,
                              const float* __restrict__ deg, long long n) {
  long long t = (long long)blockIdx.x * blockDim.x + threadIdx.x;
  if (t >= n) return;
  float dg = deg[t >> 8];
  out[t] += accum[t] / fmaxf(dg, 1.0f);
}

// ---------------------------------------------------------------------------
static inline size_t align256(size_t x) { return (x + 255) & ~(size_t)255; }

extern "C" void kernel_launch(void* const* d_in, const int* in_sizes, int n_in,
                              void* d_out, int out_size, void* d_ws,
                              size_t ws_size, hipStream_t stream) {
  const float* feat = (const float*)d_in[0];
  const float* Wself = (const float*)d_in[1];
  const float* Wneigh = (const float*)d_in[2];
  const int* src = (const int*)d_in[3];
  const int* dst = (const int*)d_in[4];
  float* out = (float*)d_out;

  const int N = in_sizes[0] / KDIM;   // 100000
  const int E = in_sizes[3];          // 1600000

  // workspace carve-up
  char* ws = (char*)d_ws;
  size_t off = 0;
  bf16_t* featb = (bf16_t*)(ws + off); off = align256(off + (size_t)N * KDIM * 2);
  bf16_t* wtS   = (bf16_t*)(ws + off); off = align256(off + (size_t)KDIM * NOUT * 2);
  bf16_t* wtN   = (bf16_t*)(ws + off); off = align256(off + (size_t)KDIM * NOUT * 2);
  float* hneigh = (float*)(ws + off);  off = align256(off + (size_t)N * NOUT * 4);
  float* svals  = (float*)(ws + off);  off = align256(off + (size_t)N * TOPK * 4);
  int*   scols  = (int*)(ws + off);    off = align256(off + (size_t)N * TOPK * 4);
  float* deg    = (float*)(ws + off);  off = align256(off + (size_t)N * 4);
  (void)ws_size; (void)n_in; (void)out_size;

  // 1) bf16 conversions (+ W transpose so B frags are contiguous 32B chunks)
  long long n8 = (long long)N * KDIM / 8;
  cvt_feat8<<<(unsigned)((n8 + 255) / 256), 256, 0, stream>>>(feat, featb, n8);
  cvt_w_transpose<<<(KDIM * NOUT) / 256, 256, 0, stream>>>(Wself, wtS);
  cvt_w_transpose<<<(KDIM * NOUT) / 256, 256, 0, stream>>>(Wneigh, wtN);

  // 2) fused dual GEMM: h_self -> d_out, h_neigh -> ws
  int Mtiles = (N + 15) / 16;
  int waves = Mtiles * 4;
  gemm_dual<<<(waves + 7) / 8, 256, 0, stream>>>(featb, wtS, wtN, out, hneigh,
                                                 N, Mtiles);

  // 3) degree reset, MaxK sparsify (also zeroes hneigh -> accumulator)
  zero_f32<<<(N + 255) / 256, 256, 0, stream>>>(deg, N);
  maxk_sparse<<<(N + 7) / 8, 256, 0, stream>>>(hneigh, svals, scols, N);

  // 4) sparse edge scatter (L2-resident accumulator) + degree count
  edge_scatter<<<(E + 7) / 8, 256, 0, stream>>>(src, dst, svals, scols, hneigh,
                                                deg, E);

  // 5) out = h_self + accum / max(deg, 1)
  long long ntot = (long long)N * NOUT;
  finalize_mean<<<(unsigned)((ntot + 255) / 256), 256, 0, stream>>>(out, hneigh,
                                                                    deg, ntot);
}
